// TransformerDecoderLayer_18897856103092
// MI455X (gfx1250) — matrix operand order repaired and checked by reading, not compiled, not run on hardware
//
#include <hip/hip_runtime.h>
#include <math.h>

// ---------------------------------------------------------------------------
// Types / helpers
// ---------------------------------------------------------------------------
typedef __attribute__((ext_vector_type(16))) __bf16 v16bf;
typedef __attribute__((ext_vector_type(8)))  float  v8f;

#define DEV static __device__ __forceinline__

DEV unsigned short f2bf(float f) {
    union { float f; unsigned u; } c; c.f = f;
    unsigned u = c.u;
    return (unsigned short)((u + 0x7FFFu + ((u >> 16) & 1u)) >> 16);   // RNE
}
DEV unsigned pack2(float a, float b) {
    return (unsigned)f2bf(a) | ((unsigned)f2bf(b) << 16);
}
// LDS byte offset of a __shared__ object (generic addr low 32 bits == LDS offset)
DEV unsigned lds_off(const void* p) { return (unsigned)(size_t)p; }

// CDNA5 async global->LDS copy, 16B per lane, tracked by ASYNCcnt
DEV void async_b128(unsigned lds_addr, unsigned goff, const void* base) {
    asm volatile("global_load_async_to_lds_b128 %0, %1, %2"
                 :: "v"(lds_addr), "v"(goff), "s"(base) : "memory");
}
DEV void wait_async0() { asm volatile("s_wait_asynccnt 0" ::: "memory"); }

union FragBF { v16bf v; unsigned u[8]; };

// ---------------------------------------------------------------------------
// fp32 -> bf16 convert (weights / one-time)
// ---------------------------------------------------------------------------
__global__ __launch_bounds__(256)
void cvt_kernel(const float* __restrict__ src, unsigned short* __restrict__ dst, int n)
{
    int i = blockIdx.x * 256 + threadIdx.x;
    int stride = gridDim.x * 256;
    for (; i < n; i += stride) dst[i] = f2bf(src[i]);
}

// ---------------------------------------------------------------------------
// Embedding * sqrt(d) + sinusoidal positional encoding  (d = 1024, S = 1024)
// ---------------------------------------------------------------------------
__global__ __launch_bounds__(256)
void embed_kernel(const int* __restrict__ toks, const float* __restrict__ emb,
                  float* __restrict__ out)
{
    const int row = blockIdx.x, tid = threadIdx.x;
    const int s = row & 1023;
    const int tok = toks[row];
    const float c = 0.00899447301f;   // ln(10000)/1024
    for (int i = 0; i < 4; i++) {
        int col = tid + i * 256;
        float e = emb[(size_t)tok * 1024 + col] * 32.0f;   // sqrt(1024)
        int ii = col >> 1;
        float den = __expf(-(float)(2 * ii) * c);
        float ang = (float)s * den;
        float pe = (col & 1) ? __cosf(ang) : __sinf(ang);
        out[(size_t)row * 1024 + col] = e + pe;
    }
}

// ---------------------------------------------------------------------------
// LayerNorm over d=1024; one block per row
// ---------------------------------------------------------------------------
template<bool OUTBF>
__global__ __launch_bounds__(256)
void ln_kernel(const float* __restrict__ x, const float* __restrict__ g,
               const float* __restrict__ bta, void* __restrict__ y)
{
    __shared__ float rs[256], rq[256];
    const int row = blockIdx.x, tid = threadIdx.x;
    float v[4]; float s = 0.f, q = 0.f;
    for (int i = 0; i < 4; i++) {
        v[i] = x[(size_t)row * 1024 + tid + i * 256];
        s += v[i]; q += v[i] * v[i];
    }
    rs[tid] = s; rq[tid] = q; __syncthreads();
    for (int off = 128; off > 0; off >>= 1) {
        if (tid < off) { rs[tid] += rs[tid + off]; rq[tid] += rq[tid + off]; }
        __syncthreads();
    }
    float mu = rs[0] * (1.0f / 1024.0f);
    float var = rq[0] * (1.0f / 1024.0f) - mu * mu;
    float r = rsqrtf(var + 1e-6f);
    for (int i = 0; i < 4; i++) {
        int col = tid + i * 256;
        float o = (v[i] - mu) * r * g[col] + bta[col];
        if (OUTBF) ((unsigned short*)y)[(size_t)row * 1024 + col] = f2bf(o);
        else       ((float*)y)[(size_t)row * 1024 + col] = o;
    }
}

// ---------------------------------------------------------------------------
// GEMM: C[M,N] = epilogue( A[M,K] @ W[K,N] + bias [+ res] )
//   Double-buffered LDS; A staged via async global->LDS (bf16 path),
//   B (transpose) and fp32-A (convert) staged via registers, committed after
//   the WMMAs so global latency overlaps compute.
//   Block 256 thr (8 waves), tile 128x128, wave tile 64x32, K-step 32.
// ---------------------------------------------------------------------------
template<bool AF32, bool OUTBF, bool RELU, bool RES>
__global__ __launch_bounds__(256)
void gemm_kernel(const void* __restrict__ Ap, const unsigned short* __restrict__ W,
                 const float* __restrict__ bias, const float* __restrict__ res,
                 void* __restrict__ Cp, int M, int N, int K)
{
    __shared__ unsigned Asl[2][128 * 20];   // rows padded to 20 dwords (16B-aligned)
    __shared__ unsigned Bsl[2][128 * 17];   // Bt[n][k], 17-dword rows

    const int tid = threadIdx.x;
    const int lane = tid & 31, wid = tid >> 5;
    const int lane16 = lane & 15, half = lane >> 4;
    const int wm = wid >> 2, wn = wid & 3;           // waves: 2 (M) x 4 (N)
    const int m0 = blockIdx.y * 128, n0 = blockIdx.x * 128;
    const int KT = K >> 5;

    v8f acc[4][2];
    for (int a = 0; a < 4; a++)
        for (int b = 0; b < 2; b++)
            for (int i = 0; i < 8; i++) acc[a][b][i] = 0.f;

    const unsigned* Wd = (const unsigned*)W;

    // ---- prologue: stage tile 0 ----
    if (AF32) {
        const float* Af = (const float*)Ap;
        for (int i = 0; i < 8; i++) {
            int idx = tid + i * 256;
            int row = idx >> 4, cp = idx & 15;
            size_t g = (size_t)(m0 + row) * K + cp * 2;
            Asl[0][row * 20 + cp] = pack2(Af[g], Af[g + 1]);
        }
    } else {
        for (int i = 0; i < 2; i++) {
            int idx = tid + i * 256;                 // 512 x 16B chunks
            int row = idx >> 2, q = idx & 3;
            async_b128(lds_off(&Asl[0][row * 20 + q * 4]),
                       (unsigned)((m0 + row) * (K << 1) + q * 16), Ap);
        }
    }
    {
        unsigned short* Bx = (unsigned short*)Bsl[0];
        for (int i = 0; i < 8; i++) {
            int idx = tid + i * 256;
            int kk = idx >> 6, np = idx & 63;
            unsigned w2 = Wd[(size_t)kk * (N >> 1) + (n0 >> 1) + np];
            int n = np * 2;
            Bx[n * 34 + kk]       = (unsigned short)(w2 & 0xFFFFu);
            Bx[(n + 1) * 34 + kk] = (unsigned short)(w2 >> 16);
        }
    }
    wait_async0();
    __syncthreads();

    for (int kt = 0; kt < KT; kt++) {
        const int cur = kt & 1, nxt = cur ^ 1;
        const int k0n = (kt + 1) * 32;
        const bool have_next = (kt + 1 < KT);

        // ---- issue next-tile global traffic (no waits here) ----
        unsigned bnext[8];
        float af0[8], af1[8];
        if (have_next) {
            if (AF32) {
                const float* Af = (const float*)Ap;
                for (int i = 0; i < 8; i++) {
                    int idx = tid + i * 256;
                    int row = idx >> 4, cp = idx & 15;
                    size_t g = (size_t)(m0 + row) * K + k0n + cp * 2;
                    af0[i] = Af[g]; af1[i] = Af[g + 1];
                }
            } else {
                for (int i = 0; i < 2; i++) {
                    int idx = tid + i * 256;
                    int row = idx >> 2, q = idx & 3;
                    async_b128(lds_off(&Asl[nxt][row * 20 + q * 4]),
                               (unsigned)((m0 + row) * (K << 1) + k0n * 2 + q * 16), Ap);
                }
            }
            for (int i = 0; i < 8; i++) {
                int idx = tid + i * 256;
                int kk = idx >> 6, np = idx & 63;
                bnext[i] = Wd[(size_t)(k0n + kk) * (N >> 1) + (n0 >> 1) + np];
            }
        }

        // ---- compute on current buffer: fragments + 8 WMMAs ----
        FragBF fa[4], fb[2];
        for (int mt = 0; mt < 4; mt++) {
            int row = wm * 64 + mt * 16 + lane16;
            for (int j = 0; j < 8; j++)
                fa[mt].u[j] = Asl[cur][row * 20 + j + half * 8];
        }
        for (int nt = 0; nt < 2; nt++) {
            int n = wn * 32 + nt * 16 + lane16;
            for (int j = 0; j < 8; j++)
                fb[nt].u[j] = Bsl[cur][n * 17 + j + half * 8];
        }
        for (int mt = 0; mt < 4; mt++)
            for (int nt = 0; nt < 2; nt++)
                acc[mt][nt] = __builtin_amdgcn_wmma_f32_16x16x32_bf16(
                    false, fa[mt].v, false, fb[nt].v, (short)0, acc[mt][nt],
                    false, false);

        // ---- commit register-staged data into next buffer ----
        if (have_next) {
            unsigned short* Bx = (unsigned short*)Bsl[nxt];
            for (int i = 0; i < 8; i++) {
                int idx = tid + i * 256;
                int kk = idx >> 6, np = idx & 63;
                int n = np * 2;
                Bx[n * 34 + kk]       = (unsigned short)(bnext[i] & 0xFFFFu);
                Bx[(n + 1) * 34 + kk] = (unsigned short)(bnext[i] >> 16);
            }
            if (AF32) {
                for (int i = 0; i < 8; i++) {
                    int idx = tid + i * 256;
                    int row = idx >> 4, cp = idx & 15;
                    Asl[nxt][row * 20 + cp] = pack2(af0[i], af1[i]);
                }
            }
        }
        wait_async0();
        __syncthreads();
    }

    // ---- epilogue ----
    for (int mt = 0; mt < 4; mt++) {
        for (int nt = 0; nt < 2; nt++) {
            int colg = n0 + wn * 32 + nt * 16 + lane16;
            float bv = bias[colg];
            for (int i = 0; i < 8; i++) {
                int rowg = m0 + wm * 64 + mt * 16 + half * 8 + i;
                float v = acc[mt][nt][i] + bv;
                if (RES)  v += res[(size_t)rowg * N + colg];
                if (RELU) v = fmaxf(v, 0.f);
                if (OUTBF) ((unsigned short*)Cp)[(size_t)rowg * N + colg] = f2bf(v);
                else       ((float*)Cp)[(size_t)rowg * N + colg] = v;
            }
        }
    }
}

// ---------------------------------------------------------------------------
// Flash attention: block = (b, h, 64 q-rows); 4 waves x 16 q rows each.
// dk = 64, H = 16, scale = 0.125. Q and K staged with async global->LDS.
// MASK: query-pad (token==1) OR causal (j > q)  ->  -1e9 (matches reference).
// ---------------------------------------------------------------------------
template<bool MASK>
__global__ __launch_bounds__(128)
void flash_kernel(const unsigned short* __restrict__ Q,
                  const unsigned short* __restrict__ Kb,
                  const unsigned short* __restrict__ Vb,
                  unsigned short* __restrict__ O,
                  const int* __restrict__ toks, int Skv)
{
    __shared__ unsigned Qs[64 * 36];     // 36-dword rows (16B aligned for async)
    __shared__ unsigned Ks[32 * 36];
    __shared__ unsigned Vt[64 * 17];     // V transposed: Vt[dk][kv]
    __shared__ unsigned Pw[64 * 17];     // P (4 waves x 16 x 32 bf16)
    __shared__ float    Sw[4 * 16 * 33];
    __shared__ float    oscl[64], lsum[64];

    const int tid = threadIdx.x;
    const int lane = tid & 31, w = tid >> 5;
    const int lane16 = lane & 15, half = lane >> 4;
    const int b = blockIdx.z, h = blockIdx.y;
    const int q0 = blockIdx.x * 64;
    const int Sq = 1024, D = 1024, DK = 64;

    const unsigned* Vd = (const unsigned*)Vb;
    unsigned short* Vt16 = (unsigned short*)Vt;
    unsigned short* Pw16 = (unsigned short*)Pw;

    // stage Q tile once (64 rows x 128B, async 16B chunks)
    for (int i = 0; i < 4; i++) {
        int idx = tid + i * 128;
        int row = idx >> 3, q = idx & 7;
        async_b128(lds_off(&Qs[row * 36 + q * 4]),
                   (unsigned)(((b * Sq + q0 + row) * D + h * DK) * 2 + q * 16), Q);
    }

    const int qg = q0 + w * 16 + lane16;           // this lane's q row (lanes<16)
    bool rowpad = false;
    if (MASK && lane < 16) rowpad = (toks[b * Sq + qg] == 1);

    float m_run = -3e38f, l_run = 0.f;
    v8f oacc[4];
    for (int nt = 0; nt < 4; nt++)
        for (int i = 0; i < 8; i++) oacc[nt][i] = 0.f;

    for (int j0 = 0; j0 < Skv; j0 += 32) {
        __syncthreads();
        // ---- stage K (async) and V (transposed, via registers) ----
        for (int i = 0; i < 2; i++) {
            int idx = tid + i * 128;
            int row = idx >> 3, q = idx & 7;
            async_b128(lds_off(&Ks[row * 36 + q * 4]),
                       (unsigned)(((b * Skv + j0 + row) * D + h * DK) * 2 + q * 16), Kb);
        }
        for (int i = 0; i < 8; i++) {
            int idx = tid + i * 128;
            int row = idx >> 5, cp = idx & 31;
            unsigned v2 = Vd[((size_t)(b * Skv + j0 + row) * D + h * DK) / 2 + cp];
            Vt16[(2 * cp) * 34 + row]     = (unsigned short)(v2 & 0xFFFFu);
            Vt16[(2 * cp + 1) * 34 + row] = (unsigned short)(v2 >> 16);
        }
        wait_async0();
        __syncthreads();

        // ---- S = Q . K^T  (16x32 per wave) ----
        v8f sacc[2];
        for (int nt = 0; nt < 2; nt++)
            for (int i = 0; i < 8; i++) sacc[nt][i] = 0.f;
        {
            const int qrow = w * 16 + lane16;
            for (int c = 0; c < 2; c++) {
                FragBF fq;
                for (int j = 0; j < 8; j++)
                    fq.u[j] = Qs[qrow * 36 + c * 16 + j + half * 8];
                for (int nt = 0; nt < 2; nt++) {
                    FragBF fk;
                    int jj = nt * 16 + lane16;
                    for (int j = 0; j < 8; j++)
                        fk.u[j] = Ks[jj * 36 + c * 16 + j + half * 8];
                    sacc[nt] = __builtin_amdgcn_wmma_f32_16x16x32_bf16(
                        false, fq.v, false, fk.v, (short)0, sacc[nt], false, false);
                }
            }
        }
        for (int nt = 0; nt < 2; nt++)
            for (int i = 0; i < 8; i++)
                Sw[w * 528 + (half * 8 + i) * 33 + nt * 16 + lane16] = sacc[nt][i];
        __syncthreads();

        // ---- online softmax: lanes 0..15 each own one q row ----
        if (lane < 16) {
            const int m = lane16;
            float sv[32];
            float mx = -3e38f;
            for (int j = 0; j < 32; j++) {
                float s = Sw[w * 528 + m * 33 + j] * 0.125f;
                if (MASK) {
                    int jgl = j0 + j;
                    if (jgl > qg || rowpad) s = -1e9f;
                }
                sv[j] = s; mx = fmaxf(mx, s);
            }
            float nm = fmaxf(m_run, mx);
            float osc = __expf(m_run - nm);
            float sum = 0.f;
            for (int j = 0; j < 32; j++) {
                float p = __expf(sv[j] - nm);
                sum += p;
                Pw16[(w * 16 + m) * 34 + j] = f2bf(p);
            }
            l_run = l_run * osc + sum;
            m_run = nm;
            oscl[w * 16 + m] = osc;
        }
        __syncthreads();

        // ---- rescale O and accumulate P . V ----
        float osc8[8];
        for (int i = 0; i < 8; i++) osc8[i] = oscl[w * 16 + half * 8 + i];
        FragBF fp;
        for (int j = 0; j < 8; j++)
            fp.u[j] = Pw[(w * 16 + lane16) * 17 + j + half * 8];
        for (int nt = 0; nt < 4; nt++) {
            for (int i = 0; i < 8; i++) oacc[nt][i] *= osc8[i];
            FragBF fv;
            int n = nt * 16 + lane16;
            for (int j = 0; j < 8; j++)
                fv.u[j] = Vt[n * 17 + j + half * 8];
            oacc[nt] = __builtin_amdgcn_wmma_f32_16x16x32_bf16(
                false, fp.v, false, fv.v, (short)0, oacc[nt], false, false);
        }
    }

    if (lane < 16) lsum[w * 16 + lane16] = l_run;
    __syncthreads();

    for (int nt = 0; nt < 4; nt++) {
        int col = h * DK + nt * 16 + lane16;
        for (int i = 0; i < 8; i++) {
            int row = q0 + w * 16 + half * 8 + i;
            float l = lsum[w * 16 + half * 8 + i];
            O[(size_t)(b * Sq + row) * D + col] = f2bf(oacc[nt][i] / l);
        }
    }
}

// ---------------------------------------------------------------------------
// Host orchestration
// ---------------------------------------------------------------------------
extern "C" void kernel_launch(void* const* d_in, const int* in_sizes, int n_in,
                              void* d_out, int out_size, void* d_ws, size_t ws_size,
                              hipStream_t stream)
{
    (void)in_sizes; (void)n_in; (void)out_size; (void)ws_size;
    const int B = 4, S = 1024, D = 1024, DFF = 4096;
    const int M = B * S;                      // 4096 rows

    const int*   toks   = (const int*)  d_in[0];
    const float* memory = (const float*)d_in[1];
    const float* emb    = (const float*)d_in[2];
    const float* ff_w1  = (const float*)d_in[3];
    const float* ff_b1  = (const float*)d_in[4];
    const float* ff_w2  = (const float*)d_in[5];
    const float* ff_b2  = (const float*)d_in[6];
    const float* ln_g   = (const float*)d_in[7];
    const float* ln_b   = (const float*)d_in[8];
    const float* sa_wq  = (const float*)d_in[9];
    const float* sa_bq  = (const float*)d_in[10];
    const float* sa_wk  = (const float*)d_in[11];
    const float* sa_bk  = (const float*)d_in[12];
    const float* sa_wv  = (const float*)d_in[13];
    const float* sa_bv  = (const float*)d_in[14];
    const float* sa_wo  = (const float*)d_in[15];
    const float* sa_bo  = (const float*)d_in[16];
    const float* xa_wq  = (const float*)d_in[17];
    const float* xa_bq  = (const float*)d_in[18];
    const float* xa_wk  = (const float*)d_in[19];
    const float* xa_bk  = (const float*)d_in[20];
    const float* xa_wv  = (const float*)d_in[21];
    const float* xa_bv  = (const float*)d_in[22];
    const float* xa_wo  = (const float*)d_in[23];
    const float* xa_bo  = (const float*)d_in[24];

    char* ws = (char*)d_ws;
    const size_t MB = (size_t)1 << 20;
    const int E = 1 << 20;                                   // 1M elems (d*d)
    unsigned short* wts = (unsigned short*)(ws);             // 16M bf16 = 32MB
    unsigned short* nbf = (unsigned short*)(ws + 32 * MB);   // LN output (bf16)
    unsigned short* qb  = (unsigned short*)(ws + 40 * MB);
    unsigned short* kb  = (unsigned short*)(ws + 48 * MB);
    unsigned short* vb  = (unsigned short*)(ws + 56 * MB);
    unsigned short* xk  = (unsigned short*)(ws + 64 * MB);   // cross K (shared wts)
    unsigned short* xv  = (unsigned short*)(ws + 72 * MB);   // cross V
    unsigned short* at  = (unsigned short*)(ws + 80 * MB);   // attention output
    unsigned short* h1  = (unsigned short*)(ws + 88 * MB);   // FFN hidden, 32MB

    // ---- one-time: weights -> bf16 ----
    cvt_kernel<<<256, 256, 0, stream>>>(sa_wq, wts + 0 * E, E);
    cvt_kernel<<<256, 256, 0, stream>>>(sa_wk, wts + 1 * E, E);
    cvt_kernel<<<256, 256, 0, stream>>>(sa_wv, wts + 2 * E, E);
    cvt_kernel<<<256, 256, 0, stream>>>(sa_wo, wts + 3 * E, E);
    cvt_kernel<<<256, 256, 0, stream>>>(xa_wq, wts + 4 * E, E);
    cvt_kernel<<<256, 256, 0, stream>>>(xa_wk, wts + 5 * E, E);
    cvt_kernel<<<256, 256, 0, stream>>>(xa_wv, wts + 6 * E, E);
    cvt_kernel<<<256, 256, 0, stream>>>(xa_wo, wts + 7 * E, E);
    cvt_kernel<<<256, 256, 0, stream>>>(ff_w1, wts + 8 * E, 4 * E);
    cvt_kernel<<<256, 256, 0, stream>>>(ff_w2, wts + 12 * E, 4 * E);

    float* out = (float*)d_out;               // fp32 residual stream lives in d_out
    embed_kernel<<<M, 256, 0, stream>>>(toks, emb, out);

    dim3 g1(D / 128,  M / 128);   // N=1024
    dim3 g2(DFF / 128, M / 128);  // N=4096
    dim3 fg(S / 64, 16, B);       // flash grid: (qtiles, heads, batch)

    // cross-attention K/V: weights shared across stacks + memory constant -> once
    gemm_kernel<true,  true,  false, false><<<g1, 256, 0, stream>>>(memory, wts + 5 * E, xa_bk, nullptr, xk, M, D, D);
    gemm_kernel<true,  true,  false, false><<<g1, 256, 0, stream>>>(memory, wts + 6 * E, xa_bv, nullptr, xv, M, D, D);

    for (int l = 0; l < 4; l++) {
        // --- self attention ---
        ln_kernel<true><<<M, 256, 0, stream>>>(out, ln_g, ln_b, nbf);
        gemm_kernel<false, true,  false, false><<<g1, 256, 0, stream>>>(nbf, wts + 0 * E, sa_bq, nullptr, qb, M, D, D);
        gemm_kernel<false, true,  false, false><<<g1, 256, 0, stream>>>(nbf, wts + 1 * E, sa_bk, nullptr, kb, M, D, D);
        gemm_kernel<false, true,  false, false><<<g1, 256, 0, stream>>>(nbf, wts + 2 * E, sa_bv, nullptr, vb, M, D, D);
        flash_kernel<true><<<fg, 128, 0, stream>>>(qb, kb, vb, at, toks, S);
        gemm_kernel<false, false, false, true ><<<g1, 256, 0, stream>>>(at, wts + 3 * E, sa_bo, out, out, M, D, D);
        // --- cross attention ---
        ln_kernel<true><<<M, 256, 0, stream>>>(out, ln_g, ln_b, nbf);
        gemm_kernel<false, true,  false, false><<<g1, 256, 0, stream>>>(nbf, wts + 4 * E, xa_bq, nullptr, qb, M, D, D);
        flash_kernel<false><<<fg, 128, 0, stream>>>(qb, xk, xv, at, nullptr, S);
        gemm_kernel<false, false, false, true ><<<g1, 256, 0, stream>>>(at, wts + 7 * E, xa_bo, out, out, M, D, D);
        // --- feed forward ---
        ln_kernel<true><<<M, 256, 0, stream>>>(out, ln_g, ln_b, nbf);
        gemm_kernel<false, true,  true,  false><<<g2, 256, 0, stream>>>(nbf, wts + 8 * E, ff_b1, nullptr, h1, M, DFF, D);
        gemm_kernel<false, false, false, true ><<<g1, 256, 0, stream>>>(h1, wts + 12 * E, ff_b2, out, out, M, D, DFF);
    }

    // final LayerNorm (in-place on d_out: rows are register-buffered before write)
    ln_kernel<false><<<M, 256, 0, stream>>>(out, ln_g, ln_b, d_out);
}
